// DySample_10213432229932
// MI455X (gfx1250) — matrix operand order repaired and checked by reading, not compiled, not run on hardware
//
#include <hip/hip_runtime.h>
#include <hip/hip_fp16.h>

typedef __attribute__((ext_vector_type(16))) _Float16 v16h;
typedef __attribute__((ext_vector_type(8)))  float    v8f;
typedef __attribute__((ext_vector_type(4)))  float    v4f;

#define Bn   8
#define CIN  64
#define Hn   256
#define Wn   256
#define SH   512
#define SW   512
#define OCH  64
#define POS  16      // base positions per workgroup
#define PIX  64      // output pixels per workgroup (POS * 4)

// padded LDS strides (bank-conflict-free for WMMA fragment access patterns)
#define XH_S 72      // f16, x tile       [16][64]
#define WL_S 72      // f16, off+scope w  [64][64]
#define WE_S 72      // f16, end_w        [64][64]
#define SA_S 72      // f16, sampled tile [64][64]
#define OF_S 68      // f32, offsets      [16][64]
#define RS_S 68      // f32, result tile  [64][64]

__device__ __forceinline__ float fast_sigmoid(float s) {
    // native v_exp + v_rcp; plenty of precision for offset prediction
    return __builtin_amdgcn_rcpf(1.0f + __expf(-s));
}

__global__ __launch_bounds__(256) void dysample_fused(
    const float* __restrict__ x,
    const float* __restrict__ offset_w,
    const float* __restrict__ offset_b,
    const float* __restrict__ scope_w,
    const float* __restrict__ end_w,
    const float* __restrict__ end_b,
    float* __restrict__ out)
{
    __shared__ _Float16 xh[POS * XH_S];
    __shared__ _Float16 wl[64 * WL_S];          // 0..31 = offset_w rows, 32..63 = scope_w rows
    __shared__ _Float16 we[64 * WE_S];
    __shared__ float    offl[POS * OF_S];       // och 0..31 = off, 32..63 = scope logits
    __shared__ _Float16 samp[PIX * SA_S];
    __shared__ __align__(16) float res[OCH * RS_S];
    __shared__ float    biasl[64];
    __shared__ float    obl[32];

    const int t    = threadIdx.x;
    const int lane = t & 31;
    const int wv   = t >> 5;
    const int w0   = blockIdx.x * POS;          // 0..255 step 16
    const int h    = blockIdx.y;                // 0..255
    const int b    = blockIdx.z;                // 0..7

    // ---------------- phase 1: stage weights + x tile to LDS (f16) ----------------
    #pragma unroll 4
    for (int i = 0; i < 16; ++i) {
        int idx = i * 256 + t;                  // 0..4095
        int och = idx >> 6, k = idx & 63;
        float v = (och < 32) ? offset_w[och * 64 + k] : scope_w[(och - 32) * 64 + k];
        wl[och * WL_S + k] = (_Float16)v;
        we[och * WE_S + k] = (_Float16)end_w[och * 64 + k];
    }
    #pragma unroll
    for (int i = 0; i < 4; ++i) {
        int idx = i * 256 + t;                  // 0..1023
        int pos = idx & 15, c = idx >> 4;
        float v = x[(((size_t)b * CIN + c) * Hn + h) * Wn + (w0 + pos)];
        xh[pos * XH_S + c] = (_Float16)v;
    }
    if (t < 64) biasl[t] = end_b[t];
    if (t < 32) obl[t]   = offset_b[t];
    __syncthreads();

    // ---------------- phase 2: offset GEMM (16 pos x 64 och), waves 0..3 ----------
    if (wv < 4) {
        const int n  = lane & 15;
        const int hi = lane >> 4;
        v8f acc = {};
        #pragma unroll
        for (int ks = 0; ks < 2; ++ks) {
            v16h af, bf;
            #pragma unroll
            for (int r = 0; r < 8; ++r) {
                int ka = ks * 32 + 2 * (r & 3) + (hi << 3) + ((r >> 2) << 4);
                af[2 * r]     = xh[n * XH_S + ka];          // A: m = lane&15
                af[2 * r + 1] = xh[n * XH_S + ka + 1];
                int kb = ks * 32 + 2 * (r & 3) + (hi << 3) + ((r >> 2) << 4);
                bf[2 * r]     = wl[(wv * 16 + n) * WL_S + kb];
                bf[2 * r + 1] = wl[(wv * 16 + n) * WL_S + kb + 1];
            }
            acc = __builtin_amdgcn_wmma_f32_16x16x32_f16(
                      false, af, false, bf, (short)0, acc, false, false);
        }
        #pragma unroll
        for (int r = 0; r < 8; ++r) {
            int m = r + (hi << 3);
            offl[m * OF_S + wv * 16 + n] = acc[r];
        }
    }
    __syncthreads();

    // ---------------- phase 3: bilinear gather, one (pixel,group) per thread ------
    {
        const int op  = t >> 2;                 // output pixel in tile, 0..63
        const int gi  = t & 3;                  // group
        const int pp  = op >> 2;                // base position
        const int sub = op & 3;
        const int ay  = sub >> 1;
        const int bx  = sub & 1;
        const int j   = gi * 4 + sub;           // g*s2 + a*s + b

        float offx = offl[pp * OF_S + j]      + obl[j];
        float scpx = offl[pp * OF_S + 32 + j];
        float offy = offl[pp * OF_S + 16 + j] + obl[16 + j];
        float scpy = offl[pp * OF_S + 48 + j];
        float dx   = offx * fast_sigmoid(scpx) * 0.5f + (bx ? 0.25f : -0.25f);
        float dy   = offy * fast_sigmoid(scpy) * 0.5f + (ay ? 0.25f : -0.25f);
        float ix   = fminf(fmaxf((float)(w0 + pp) + dx, 0.0f), (float)(Wn - 1));
        float iy   = fminf(fmaxf((float)h        + dy, 0.0f), (float)(Hn - 1));
        float fx = floorf(ix), fy = floorf(iy);
        float wx = ix - fx,    wy = iy - fy;
        int x0 = min((int)fx, Wn - 1);
        int y0 = min((int)fy, Hn - 1);
        int x1 = min(x0 + 1, Wn - 1);
        int y1 = min(y0 + 1, Hn - 1);
        float w00 = (1.0f - wx) * (1.0f - wy);
        float w01 = wx * (1.0f - wy);
        float w10 = (1.0f - wx) * wy;
        float w11 = wx * wy;
        const float* xb = x + ((size_t)b * CIN + gi * 16) * (size_t)(Hn * Wn);
        size_t o00 = (size_t)y0 * Wn + x0;
        size_t o01 = (size_t)y0 * Wn + x1;
        size_t o10 = (size_t)y1 * Wn + x0;
        size_t o11 = (size_t)y1 * Wn + x1;
        #pragma unroll 4
        for (int cc = 0; cc < 16; ++cc) {
            const float* p = xb + (size_t)cc * (Hn * Wn);   // L2-resident gathers
            float v = p[o00] * w00 + p[o01] * w01 + p[o10] * w10 + p[o11] * w11;
            samp[op * SA_S + gi * 16 + cc] = (_Float16)v;
        }
    }
    __syncthreads();

    // ---------------- phase 4: output GEMM 64x64x64, 16 WMMA tiles over 8 waves ---
    {
        const int mt  = wv & 3;
        const int nt0 = wv >> 2;                // 0 or 1; this wave does nt0 and nt0+2
        const int n   = lane & 15;
        const int hi  = lane >> 4;

        v16h afr[2];
        #pragma unroll
        for (int ks = 0; ks < 2; ++ks) {
            #pragma unroll
            for (int r = 0; r < 8; ++r) {
                int ka = ks * 32 + 2 * (r & 3) + (hi << 3) + ((r >> 2) << 4);
                afr[ks][2 * r]     = samp[(mt * 16 + n) * SA_S + ka];
                afr[ks][2 * r + 1] = samp[(mt * 16 + n) * SA_S + ka + 1];
            }
        }
        #pragma unroll
        for (int q = 0; q < 2; ++q) {
            int nt = nt0 + q * 2;
            v8f acc = {};
            #pragma unroll
            for (int ks = 0; ks < 2; ++ks) {
                v16h bf;
                #pragma unroll
                for (int r = 0; r < 8; ++r) {
                    int kb = ks * 32 + 2 * (r & 3) + (hi << 3) + ((r >> 2) << 4);
                    bf[2 * r]     = we[(nt * 16 + n) * WE_S + kb];
                    bf[2 * r + 1] = we[(nt * 16 + n) * WE_S + kb + 1];
                }
                acc = __builtin_amdgcn_wmma_f32_16x16x32_f16(
                          false, afr[ks], false, bf, (short)0, acc, false, false);
            }
            float bias = biasl[nt * 16 + n];
            #pragma unroll
            for (int r = 0; r < 8; ++r) {
                int m   = mt * 16 + r + (hi << 3);          // pixel index
                int pp  = m >> 2, sub = m & 3;
                int slot = (sub >> 1) * 32 + pp * 2 + (sub & 1);  // (a, ox-contig)
                res[(nt * 16 + n) * RS_S + slot] = acc[r] + bias;
            }
        }
    }
    __syncthreads();

    // ---------------- phase 5: coalesced non-temporal float4 stores ---------------
    #pragma unroll
    for (int i = 0; i < 4; ++i) {
        int f   = i * 256 + t;                  // 0..1023 float4s
        int q   = f & 7;                        // float4 within ox-row of 32
        int a   = (f >> 3) & 1;
        int och = f >> 4;
        v4f v = *(const v4f*)&res[och * RS_S + a * 32 + q * 4];
        size_t o = (((size_t)b * OCH + och) * SH + (size_t)(2 * h + a)) * SW
                 + (size_t)(2 * w0 + q * 4);
        __builtin_nontemporal_store(v, (v4f*)(out + o));    // don't evict x from L2
    }
}

extern "C" void kernel_launch(void* const* d_in, const int* in_sizes, int n_in,
                              void* d_out, int out_size, void* d_ws, size_t ws_size,
                              hipStream_t stream) {
    const float* x        = (const float*)d_in[0];
    const float* offset_w = (const float*)d_in[1];
    const float* offset_b = (const float*)d_in[2];
    const float* scope_w  = (const float*)d_in[3];
    const float* end_w    = (const float*)d_in[4];
    const float* end_b    = (const float*)d_in[5];
    float* out = (float*)d_out;
    (void)in_sizes; (void)n_in; (void)out_size; (void)d_ws; (void)ws_size;

    dim3 grid(Wn / POS, Hn, Bn);                // 16 x 256 x 8 workgroups
    dysample_fused<<<grid, 256, 0, stream>>>(x, offset_w, offset_b, scope_w,
                                             end_w, end_b, out);
}
